// SoftMoELayerWrapper_3719441678625
// MI455X (gfx1250) — matrix operand order repaired and checked by reading, not compiled, not run on hardware
//
#include <hip/hip_runtime.h>
#include <math.h>

// ---------------- constants ----------------
#define Bb 4
#define Nn 1024
#define Dd 1024
#define Ee 32
#define Hh 4096
#define LN_EPS 1e-5f
#define PSIM_EPS 1e-9f
#define LDP 129   // LDS tile pitch in floats (32*LDP*4 B per buffer; pad kills bank conflicts)

typedef __attribute__((ext_vector_type(16))) __bf16 v16bf;
typedef __attribute__((ext_vector_type(8)))  __bf16 v8bf;
typedef __attribute__((ext_vector_type(8)))  float  v8f;

// ---------------- WMMA operand loaders (wave32; element i of lane half sel holds k = 16*sel + i, same in A and B) ----------------
__device__ inline v16bf load_a16(const __bf16* base, size_t ld, int row, int k0, int sel) {
    const __bf16* p = base + (size_t)row * ld + (size_t)(k0 + sel * 16);
    v8bf lo = *(const v8bf*)p;
    v8bf hi = *(const v8bf*)(p + 8);
    v16bf r;
    #pragma unroll
    for (int i = 0; i < 8; ++i) { r[i] = lo[i]; r[i + 8] = hi[i]; }
    return r;
}

__device__ inline v16bf load_a16_guard(const __bf16* base, size_t rowstride,
                                       int row, int rows, int k0, int sel) {
    v16bf r;
    if (row < rows) {
        const __bf16* p = base + (size_t)row * rowstride + (size_t)(k0 + sel * 16);
        v8bf lo = *(const v8bf*)p;
        v8bf hi = *(const v8bf*)(p + 8);
        #pragma unroll
        for (int i = 0; i < 8; ++i) { r[i] = lo[i]; r[i + 8] = hi[i]; }
    } else {
        #pragma unroll
        for (int i = 0; i < 16; ++i) r[i] = (__bf16)0.0f;
    }
    return r;
}

__device__ inline v16bf load_bT_contig(const __bf16* base, size_t ld, int col, int k0, int sel) {
    const __bf16* p = base + (size_t)col * ld + (size_t)(k0 + sel * 16);
    return *(const v16bf*)p;   // 32-byte aligned
}

__device__ inline v16bf load_b_strided_bf16(const __bf16* base, size_t ld, int col, int k0, int sel) {
    const __bf16* p = base + (size_t)(k0 + sel * 16) * ld + (size_t)col;
    v16bf r;
    #pragma unroll
    for (int i = 0; i < 16; ++i) r[i] = p[(size_t)i * ld];
    return r;
}

// B-operand (32x16, K-major per lane) gathered from the LDS-staged fp32 tile.
__device__ inline v16bf b_from_lds(const float* tile, int col, int sel) {
    const float* p = tile + sel * 16 * LDP + col;
    v16bf r;
    #pragma unroll
    for (int i = 0; i < 16; ++i) r[i] = (__bf16)p[i * LDP];
    return r;
}

#define WMMA_BF16(a, b, c) \
    __builtin_amdgcn_wmma_f32_16x16x32_bf16(false, (a), false, (b), (short)0, (c), false, false)

// ---------------- async global -> LDS staging (gfx1250, ASYNCcnt-tracked) ----------------
// Stage a 32 x 128 fp32 tile (rows k0..k0+31, cols nbase..nbase+127) into `buf` with
// fully coalesced 128-bit async copies: 256 threads x 4 issues x 16 B.
__device__ inline void stage_tile_async(const float* Wsrc, size_t ldW, int k0, int nbase,
                                        float* buf, int t) {
    #pragma unroll
    for (int p = 0; p < 4; ++p) {
        int idx = p * 256 + t;
        int row = idx >> 5, grp = idx & 31;
        unsigned lds_off = (unsigned)(size_t)(buf + row * LDP + grp * 4);
        const float* g = Wsrc + (size_t)(k0 + row) * ldW + (size_t)(nbase + grp * 4);
        asm volatile("global_load_async_to_lds_b128 %0, %1, off"
                     :: "v"(lds_off), "v"(g) : "memory");
    }
}
__device__ inline void wait_async0() {
    asm volatile("s_wait_asynccnt 0x0" ::: "memory");
}

// ---------------- block reductions (blockDim == 256) ----------------
__device__ inline float blk_sum(float v, float* red) {
    int t = threadIdx.x;
    red[t] = v; __syncthreads();
    for (int s = 128; s > 0; s >>= 1) { if (t < s) red[t] += red[t + s]; __syncthreads(); }
    float r = red[0]; __syncthreads();
    return r;
}
__device__ inline float blk_max(float v, float* red) {
    int t = threadIdx.x;
    red[t] = v; __syncthreads();
    for (int s = 128; s > 0; s >>= 1) { if (t < s) red[t] = fmaxf(red[t], red[t + s]); __syncthreads(); }
    float r = red[0]; __syncthreads();
    return r;
}

// ---------------- K1: q = LayerNorm(phi*qg+qb)*lg+lb, *scale -> bf16 ----------------
__global__ __launch_bounds__(256) void q_prep(const float* phi, const float* qg, const float* qb,
                                              const float* lg, const float* lb, const float* scale,
                                              unsigned short* q_out) {
    __shared__ float red[256];
    int e = blockIdx.x, t = threadIdx.x;
    const float* row = phi + (size_t)e * Dd;
    float s = 0.f;
    for (int n = t; n < Dd; n += 256) s += row[n] * qg[n] + qb[n];
    float mean = blk_sum(s, red) * (1.0f / Dd);
    float v = 0.f;
    for (int n = t; n < Dd; n += 256) { float d = row[n] * qg[n] + qb[n] - mean; v += d * d; }
    float var = blk_sum(v, red) * (1.0f / Dd);
    float rstd = rsqrtf(var + LN_EPS);
    float sc = scale[0];
    __bf16* qo = (__bf16*)q_out;
    for (int n = t; n < Dd; n += 256) {
        float y = ((row[n] * qg[n] + qb[n] - mean) * rstd * lg[n] + lb[n]) * sc;
        qo[(size_t)e * Dd + n] = (__bf16)y;
    }
}

// ---------------- K2: k = x*kg+kb -> bf16 ; x -> bf16 ----------------
__global__ __launch_bounds__(256) void kx_prep(const float* x, const float* kg, const float* kb,
                                               unsigned short* k_out, unsigned short* x_out) {
    size_t i = (size_t)blockIdx.x * 256 + threadIdx.x;
    if (i >= (size_t)Bb * Nn * Dd) return;
    int d = (int)(i & (Dd - 1));
    float xv = x[i];
    ((__bf16*)k_out)[i] = (__bf16)(xv * kg[d] + kb[d]);
    ((__bf16*)x_out)[i] = (__bf16)xv;
}

// ---------------- K3: logits[4096,32] = k[4096,1024] @ q[32,1024]^T ----------------
__global__ __launch_bounds__(256) void logits_gemm(const unsigned short* kbp, const unsigned short* qbp,
                                                   float* logits) {
    const __bf16* kb = (const __bf16*)kbp;
    const __bf16* qb = (const __bf16*)qbp;
    int wave = blockIdx.x * 8 + (threadIdx.x >> 5);
    int lane = threadIdx.x & 31;
    int mt = wave >> 1, nt = wave & 1;
    int m0 = mt * 16, n0 = nt * 16;
    int sel = lane >> 4, mr = lane & 15;
    v8f c = {};
    for (int k0 = 0; k0 < Dd; k0 += 32) {
        v16bf a = load_a16(kb, Dd, m0 + mr, k0, sel);
        v16bf b = load_bT_contig(qb, Dd, n0 + mr, k0, sel);
        c = WMMA_BF16(a, b, c);
    }
    int col = n0 + mr;
    #pragma unroll
    for (int i = 0; i < 8; ++i)
        logits[(size_t)(m0 + i + sel * 8) * Ee + col] = c[i];
}

// ---------------- K4: softmax over N per (b,e); write dispatch^T bf16 + dw term ----------------
__global__ __launch_bounds__(256) void dispatch_softmax(const float* logits, unsigned short* dT,
                                                        float* dwterm) {
    __shared__ float red[256];
    int b = blockIdx.x >> 5, e = blockIdx.x & 31, t = threadIdx.x;
    float m = -1e30f;
    for (int n = t; n < Nn; n += 256) m = fmaxf(m, logits[((size_t)b * Nn + n) * Ee + e]);
    m = blk_max(m, red);
    float s = 0.f;
    for (int n = t; n < Nn; n += 256) s += expf(logits[((size_t)b * Nn + n) * Ee + e] - m);
    s = blk_sum(s, red);
    float inv = 1.0f / s;
    __bf16* dTo = (__bf16*)dT;
    float sq = 0.f, sp = 0.f;
    for (int n = t; n < Nn; n += 256) {
        float p = expf(logits[((size_t)b * Nn + n) * Ee + e] - m) * inv;
        dTo[((size_t)b * Ee + e) * Nn + n] = (__bf16)p;
        sq += p * p; sp += p;
    }
    sq = blk_sum(sq, red);
    sp = blk_sum(sp, red);
    if (t == 0) {
        float sdn = sp * rsqrtf(sq + PSIM_EPS);
        dwterm[b * Ee + e] = sdn * sdn;
    }
}

// ---------------- K5: softmax over 32 slots per token; bf16 combine + cw terms ----------------
__global__ __launch_bounds__(256) void combine_softmax(const float* logits, unsigned short* cbo,
                                                       float* s1sq, float* diagterm) {
    int r = blockIdx.x * 256 + threadIdx.x;
    if (r >= Bb * Nn) return;
    const float* row = logits + (size_t)r * Ee;
    float m = -1e30f;
    for (int i = 0; i < Ee; ++i) m = fmaxf(m, row[i]);
    float p[Ee], s = 0.f;
    for (int i = 0; i < Ee; ++i) { p[i] = expf(row[i] - m); s += p[i]; }
    float inv = 1.0f / s, sq = 0.f;
    for (int i = 0; i < Ee; ++i) { p[i] *= inv; sq += p[i] * p[i]; }
    float rn = rsqrtf(sq + PSIM_EPS);
    __bf16* cb = (__bf16*)cbo;
    float S1 = 0.f;
    for (int i = 0; i < Ee; ++i) { S1 += p[i] * rn; cb[(size_t)r * Ee + i] = (__bf16)p[i]; }
    s1sq[r] = S1 * S1;
    int b = r >> 10, n = r & (Nn - 1);
    if (n < Ee) diagterm[b * Ee + n] = p[n] * rn * S1;
}

// ---------------- K6: slots[b,32,1024] = dT[b,32,1024] @ x_bf[b,1024,1024] ----------------
__global__ __launch_bounds__(256) void slots_gemm(const unsigned short* dTp, const unsigned short* xbp,
                                                  unsigned short* slotsp) {
    const __bf16* dT = (const __bf16*)dTp;
    const __bf16* xb = (const __bf16*)xbp;
    __bf16* slots = (__bf16*)slotsp;
    int wave = blockIdx.x * 8 + (threadIdx.x >> 5);
    int lane = threadIdx.x & 31;
    int b = wave >> 7, rem = wave & 127;
    int et = rem >> 6, dt = rem & 63;
    int m0 = et * 16, d0 = dt * 16;
    int sel = lane >> 4, mr = lane & 15;
    const __bf16* A = dT + (size_t)b * Ee * Nn;
    const __bf16* Bm = xb + (size_t)b * Nn * Dd;
    v8f c = {};
    for (int k0 = 0; k0 < Nn; k0 += 32) {
        v16bf a = load_a16(A, Nn, m0 + mr, k0, sel);
        v16bf bb = load_b_strided_bf16(Bm, Dd, d0 + mr, k0, sel);
        c = WMMA_BF16(a, bb, c);
    }
    int col = d0 + mr;
    #pragma unroll
    for (int i = 0; i < 8; ++i)
        slots[((size_t)b * Ee + m0 + i + sel * 8) * Dd + col] = (__bf16)c[i];
}

// ---------------- K7: h = gelu(slots @ w1 + b1); w1 tiles async-staged in LDS ----------------
__global__ __launch_bounds__(256) void mlp1_gemm(const unsigned short* slotsp, const float* w1,
                                                 const float* b1, unsigned short* hp) {
    __shared__ float tile[2][32 * LDP];
    const __bf16* slots = (const __bf16*)slotsp;
    __bf16* hb = (__bf16*)hp;
    int t = threadIdx.x;
    int lane = t & 31, w = t >> 5;
    int e = blockIdx.x >> 5;            // 32 strips of 128 columns over H
    int strip = blockIdx.x & 31;
    int nbase = strip * 128;
    int sel = lane >> 4, mr = lane & 15;
    int colw = w * 16 + mr;
    const __bf16* A = slots + (size_t)e * Dd;
    const float* W = w1 + (size_t)e * Dd * Hh;

    stage_tile_async(W, Hh, 0, nbase, tile[0], t);
    wait_async0();
    __syncthreads();

    v8f c = {};
    const int NK = Dd / 32;
    for (int kb = 0; kb < NK; ++kb) {
        int cur = kb & 1;
        if (kb + 1 < NK)
            stage_tile_async(W, Hh, (kb + 1) * 32, nbase, tile[cur ^ 1], t);
        v16bf a = load_a16_guard(A, (size_t)Ee * Dd, mr, Bb, kb * 32, sel);
        v16bf bb = b_from_lds(tile[cur], colw, sel);
        c = WMMA_BF16(a, bb, c);
        wait_async0();
        __syncthreads();
    }
    int col = nbase + colw;
    float bias = b1[(size_t)e * Hh + col];
    #pragma unroll
    for (int i = 0; i < 8; ++i) {
        int row = i + sel * 8;
        if (row < Bb) {
            float v = c[i] + bias;
            float g = 0.5f * v * (1.0f + erff(v * 0.70710678118654752f));
            hb[((size_t)row * Ee + e) * Hh + col] = (__bf16)g;
        }
    }
}

// ---------------- K8: expert_out = h @ w2 + b2; w2 tiles async-staged in LDS ----------------
__global__ __launch_bounds__(256) void mlp2_gemm(const unsigned short* hp, const float* w2,
                                                 const float* b2, unsigned short* eop) {
    __shared__ float tile[2][32 * LDP];
    const __bf16* hb = (const __bf16*)hp;
    __bf16* eo = (__bf16*)eop;
    int t = threadIdx.x;
    int lane = t & 31, w = t >> 5;
    int e = blockIdx.x >> 3;            // 8 strips of 128 columns over D
    int strip = blockIdx.x & 7;
    int nbase = strip * 128;
    int sel = lane >> 4, mr = lane & 15;
    int colw = w * 16 + mr;
    const __bf16* A = hb + (size_t)e * Hh;
    const float* W = w2 + (size_t)e * Hh * Dd;

    stage_tile_async(W, Dd, 0, nbase, tile[0], t);
    wait_async0();
    __syncthreads();

    v8f c = {};
    const int NK = Hh / 32;
    for (int kb = 0; kb < NK; ++kb) {
        int cur = kb & 1;
        if (kb + 1 < NK)
            stage_tile_async(W, Dd, (kb + 1) * 32, nbase, tile[cur ^ 1], t);
        v16bf a = load_a16_guard(A, (size_t)Ee * Hh, mr, Bb, kb * 32, sel);
        v16bf bb = b_from_lds(tile[cur], colw, sel);
        c = WMMA_BF16(a, bb, c);
        wait_async0();
        __syncthreads();
    }
    int col = nbase + colw;
    float bias = b2[(size_t)e * Dd + col];
    #pragma unroll
    for (int i = 0; i < 8; ++i) {
        int row = i + sel * 8;
        if (row < Bb)
            eo[((size_t)row * Ee + e) * Dd + col] = (__bf16)(c[i] + bias);
    }
}

// ---------------- K9: out[b,1024,1024] = combine[b,1024,32] @ eo[b,32,1024] ----------------
__global__ __launch_bounds__(256) void combine_gemm(const unsigned short* cbp, const unsigned short* eop,
                                                    float* out) {
    const __bf16* cb = (const __bf16*)cbp;
    const __bf16* eo = (const __bf16*)eop;
    int wave = blockIdx.x * 8 + (threadIdx.x >> 5);
    int lane = threadIdx.x & 31;
    int b = wave >> 12, rem = wave & 4095;
    int mt = rem >> 6, dt = rem & 63;
    int m0 = mt * 16, d0 = dt * 16;
    int sel = lane >> 4, mr = lane & 15;
    const __bf16* A = cb + (size_t)b * Nn * Ee;
    const __bf16* Bm = eo + (size_t)b * Ee * Dd;
    v16bf a = load_a16(A, Ee, m0 + mr, 0, sel);
    v16bf bb = load_b_strided_bf16(Bm, Dd, d0 + mr, 0, sel);
    v8f c = {};
    c = WMMA_BF16(a, bb, c);
    int col = d0 + mr;
    #pragma unroll
    for (int i = 0; i < 8; ++i)
        out[(size_t)b * Nn * Dd + (size_t)(m0 + i + sel * 8) * Dd + col] = c[i];
}

// ---------------- K10: deterministic metric reduction ----------------
__global__ __launch_bounds__(256) void finalize_metrics(const float* s1sq, const float* diagterm,
                                                        const float* dwterm, float* outm) {
    __shared__ float red[256];
    int t = threadIdx.x;
    float a = 0.f;
    for (int i = t; i < Bb * Nn; i += 256) a += s1sq[i];
    float tot = blk_sum(a, red);
    float d = 0.f;
    for (int i = t; i < Bb * Ee; i += 256) d += diagterm[i];
    float dg = blk_sum(d, red);
    float w = 0.f;
    for (int i = t; i < Bb * Ee; i += 256) w += dwterm[i];
    float wtot = blk_sum(w, red);
    if (t == 0) {
        float w0 = dwterm[0] + dwterm[Ee] + dwterm[2 * Ee] + dwterm[3 * Ee];
        outm[0] = (tot - dg) / ((float)Bb * Nn * (Nn - 1));
        outm[1] = (wtot - w0) / ((float)Bb * Ee * 1 * (Ee * 1 - 1));
    }
}

// ---------------- host launcher ----------------
extern "C" void kernel_launch(void* const* d_in, const int* in_sizes, int n_in,
                              void* d_out, int out_size, void* d_ws, size_t ws_size,
                              hipStream_t stream) {
    const float* x   = (const float*)d_in[0];
    const float* phi = (const float*)d_in[1];
    const float* kg  = (const float*)d_in[2];
    const float* kb  = (const float*)d_in[3];
    const float* qg  = (const float*)d_in[4];
    const float* qb  = (const float*)d_in[5];
    const float* lg  = (const float*)d_in[6];
    const float* lb  = (const float*)d_in[7];
    const float* sc  = (const float*)d_in[8];
    const float* w1  = (const float*)d_in[9];
    const float* b1  = (const float*)d_in[10];
    const float* w2  = (const float*)d_in[11];
    const float* b2  = (const float*)d_in[12];
    float* out = (float*)d_out;

    char* ws = (char*)d_ws;
    size_t off = 0;
    unsigned short* q_bf  = (unsigned short*)(ws + off); off += (size_t)Ee * Dd * 2;
    unsigned short* k_bf  = (unsigned short*)(ws + off); off += (size_t)Bb * Nn * Dd * 2;
    unsigned short* x_bf  = (unsigned short*)(ws + off); off += (size_t)Bb * Nn * Dd * 2;
    float*          logit = (float*)(ws + off);          off += (size_t)Bb * Nn * Ee * 4;
    unsigned short* dT    = (unsigned short*)(ws + off); off += (size_t)Bb * Ee * Nn * 2;
    float*          dwt   = (float*)(ws + off);          off += (size_t)Bb * Ee * 4;
    off = (off + 511) & ~(size_t)511;
    unsigned short* comb  = (unsigned short*)(ws + off); off += (size_t)Bb * Nn * Ee * 2;
    float*          s1sq  = (float*)(ws + off);          off += (size_t)Bb * Nn * 4;
    float*          diag  = (float*)(ws + off);          off += (size_t)Bb * Ee * 4;
    off = (off + 511) & ~(size_t)511;
    unsigned short* slots = (unsigned short*)(ws + off); off += (size_t)Bb * Ee * Dd * 2;
    unsigned short* hbuf  = (unsigned short*)(ws + off); off += (size_t)Bb * Ee * Hh * 2;
    unsigned short* eo    = (unsigned short*)(ws + off); off += (size_t)Bb * Ee * Dd * 2;

    q_prep<<<Ee, 256, 0, stream>>>(phi, qg, qb, lg, lb, sc, q_bf);
    kx_prep<<<(Bb * Nn * Dd) / 256, 256, 0, stream>>>(x, kg, kb, k_bf, x_bf);
    logits_gemm<<<64, 256, 0, stream>>>(k_bf, q_bf, logit);          // 512 waves
    dispatch_softmax<<<Bb * Ee, 256, 0, stream>>>(logit, dT, dwt);
    combine_softmax<<<(Bb * Nn) / 256, 256, 0, stream>>>(logit, comb, s1sq, diag);
    slots_gemm<<<64, 256, 0, stream>>>(dT, x_bf, slots);             // 512 waves
    mlp1_gemm<<<Ee * 32, 256, 0, stream>>>(slots, w1, b1, hbuf);     // 1024 blocks
    mlp2_gemm<<<Ee * 8, 256, 0, stream>>>(hbuf, w2, b2, eo);         // 256 blocks
    combine_gemm<<<2048, 256, 0, stream>>>(comb, eo, out);           // 16384 waves
    finalize_metrics<<<1, 256, 0, stream>>>(s1sq, diag, dwt, out + (size_t)Bb * Nn * Dd);
}
